// _MultiHeadAttentionQuantum_65481071402753
// MI455X (gfx1250) — matrix operand order repaired and checked by reading, not compiled, not run on hardware
//
#include <hip/hip_runtime.h>
#include <math.h>

// Problem constants (from reference)
#define EMBED 1024
#define TSEQ  1024
#define BATCH 2
#define HEADS 64
#define DK    16
#define BT    (BATCH*TSEQ)   // 2048 rows in the big GEMMs

typedef __attribute__((ext_vector_type(16))) _Float16 v16h;
typedef __attribute__((ext_vector_type(8)))  _Float16 v8h;
typedef __attribute__((ext_vector_type(4)))  _Float16 v4h;
typedef __attribute__((ext_vector_type(8)))  float    v8f;
typedef __attribute__((ext_vector_type(4)))  float    v4f;

// ---- gfx1250 async global->LDS staging (guarded; fallback = sync staging) --
#ifndef HAVE_ASYNC_LDS
#if defined(__has_builtin)
#if __has_builtin(__builtin_amdgcn_global_load_async_to_lds_b128) && \
    __has_builtin(__builtin_amdgcn_s_wait_asynccnt)
#define HAVE_ASYNC_LDS 1
#endif
#endif
#endif
#ifndef HAVE_ASYNC_LDS
#define HAVE_ASYNC_LDS 0
#endif

// Builtin signature (from compiler diagnostic):
//   void __builtin_amdgcn_global_load_async_to_lds_b128(
//        int4 AS(1)* gsrc, int4 AS(3)* lds, imm int offset, imm int cpol)
typedef int b128_t __attribute__((vector_size(16)));
typedef __attribute__((address_space(1))) b128_t* gbl_b128_t;
typedef __attribute__((address_space(3))) b128_t* lds_b128_t;

static __device__ __forceinline__ void stage16(void* lds_dst, const void* gsrc) {
#if HAVE_ASYNC_LDS
  __builtin_amdgcn_global_load_async_to_lds_b128(
      (gbl_b128_t)(uintptr_t)gsrc, (lds_b128_t)(unsigned)(uintptr_t)lds_dst,
      0, 0);
#else
  *(v8h*)lds_dst = *(const v8h*)gsrc;
#endif
}

static __device__ __forceinline__ v16h hcat(v8h lo, v8h hi) {
  v16h r;
#pragma unroll
  for (int i = 0; i < 8; ++i) { r[i] = lo[i]; r[i + 8] = hi[i]; }
  return r;
}
static __device__ __forceinline__ v8h hz8() {
  v8h r;
#pragma unroll
  for (int i = 0; i < 8; ++i) r[i] = (_Float16)0.f;
  return r;
}
static __device__ __forceinline__ v8f fz8() {
  v8f r;
#pragma unroll
  for (int i = 0; i < 8; ++i) r[i] = 0.f;
  return r;
}
static __device__ __forceinline__ v8f wmma16(v16h a, v16h b, v8f c) {
  // D = A(16x32 f16) * B(32x16 f16) + C(16x16 f32)
  return __builtin_amdgcn_wmma_f32_16x16x32_f16(false, a, false, b, (short)0, c,
                                                false, false);
}

// ---------------------------------------------------------------------------
// fp32 -> f16 conversion, 4 elements/thread
// ---------------------------------------------------------------------------
__global__ void k_cvt(const float* __restrict__ src, _Float16* __restrict__ dst,
                      int n4) {
  int i = blockIdx.x * blockDim.x + threadIdx.x;
  if (i < n4) {
    v4f f = ((const v4f*)src)[i];
    v4h h;
#pragma unroll
    for (int j = 0; j < 4; ++j) h[j] = (_Float16)f[j];
    ((v4h*)dst)[i] = h;
  }
}

// ---------------------------------------------------------------------------
// Shared GEMM core: one block (8 waves) computes a 128(M) x 64(N) tile of
// X[BT,EMBED] @ W[EMBED,EMBED]^T. The 64x32 f16 weight B-tile per k-step is
// staged to LDS (async double-buffered when available) and shared by all 8
// waves; A fragments stream straight from global (L2-resident).
// Each wave owns a 16x64 strip -> acc[4] of 16x16 f32.
// ---------------------------------------------------------------------------
#if HAVE_ASYNC_LDS
#define WAIT_ASYNC(n) __builtin_amdgcn_s_wait_asynccnt(n)
#else
#define WAIT_ASYNC(n) ((void)0)
#endif

#define GEMM_CORE(X, W, bt, acc)                                              \
  const int tid = threadIdx.x;                                                \
  const int lane = tid & 31;                                                  \
  const int wave = tid >> 5;                                                  \
  const int bm = (int)blockIdx.x >> 4;        /* 16 M-blocks of 128 rows  */  \
  const int bn = ((int)blockIdx.x & 15) * 64; /* 16 N-strips of 64 cols   */  \
  const int r = lane & 15, h = lane >> 4;                                     \
  const int mrow = bm * 128 + wave * 16 + r;                                  \
  const int srow = tid >> 2;       /* staging: 64 rows x 4 chunks of 16B */   \
  const int scol = (tid & 3) * 8;  /* halves 0,8,16,24 within 32-k chunk */   \
  const _Float16* gW = (W) + (bn + srow) * EMBED + scol;                      \
  v8f acc[4];                                                                 \
  _Pragma("unroll") for (int j = 0; j < 4; ++j) acc[j] = fz8();               \
  const int NK = EMBED / 32;                                                  \
  if (HAVE_ASYNC_LDS) stage16(&bt[0][srow * 32 + scol], gW);                  \
  for (int i = 0; i < NK; ++i) {                                              \
    const int k0 = i * 32;                                                    \
    const _Float16* Bt;                                                       \
    if (HAVE_ASYNC_LDS) {                                                     \
      if (i + 1 < NK) {                                                       \
        stage16(&bt[(i + 1) & 1][srow * 32 + scol], gW + k0 + 32);            \
        WAIT_ASYNC(1);                                                        \
      } else {                                                                \
        WAIT_ASYNC(0);                                                        \
      }                                                                       \
      __builtin_amdgcn_s_barrier();                                           \
      Bt = (const _Float16*)&bt[i & 1][0];                                    \
    } else {                                                                  \
      stage16(&bt[0][srow * 32 + scol], gW + k0);                             \
      __syncthreads();                                                        \
      Bt = (const _Float16*)&bt[0][0];                                        \
    }                                                                         \
    const _Float16* ap = (X) + mrow * EMBED + k0;                             \
    __builtin_prefetch(ap + 64, 0, 1); /* next-next A chunk */                \
    v16h a = hcat(*(const v8h*)(ap + 8 * h), *(const v8h*)(ap + 16 + 8 * h)); \
    _Pragma("unroll") for (int j = 0; j < 4; ++j) {                           \
      const _Float16* wp = Bt + (j * 16 + r) * 32 + 8 * h;                    \
      v16h b = hcat(*(const v8h*)wp, *(const v8h*)(wp + 16));                 \
      acc[j] = wmma16(a, b, acc[j]);                                          \
    }                                                                         \
    if (HAVE_ASYNC_LDS) {                                                     \
      asm volatile("s_wait_dscnt 0" ::: "memory");                            \
      __builtin_amdgcn_s_barrier();                                           \
    } else {                                                                  \
      __syncthreads();                                                        \
    }                                                                         \
  }

// ---------------------------------------------------------------------------
// Projection GEMM: P = cos(X @ W^T + bias + theta), stored f16.
// vmode==0: out[b,h,t,d]   (Q, K)
// vmode==1: out[b,h,d,t]   (V transposed for contiguous P@V B-tiles)
// ---------------------------------------------------------------------------
__global__ void k_proj(const _Float16* __restrict__ X,
                       const _Float16* __restrict__ W,
                       const float* __restrict__ bias,
                       const float* __restrict__ theta_p,
                       _Float16* __restrict__ out, int vmode) {
  __shared__ __align__(16) _Float16 bt[2][64 * 32];  // 8 KiB double buffer
  GEMM_CORE(X, W, bt, acc)

  const float th = *theta_p;
#pragma unroll
  for (int j = 0; j < 4; ++j) {
    const int n = bn + j * 16 + r;             // C layout: N = lane%16
    const float bb = bias[n] + th;
    const int head = n >> 4, d = n & 15;
#pragma unroll
    for (int i = 0; i < 8; ++i) {
      const int m = bm * 128 + wave * 16 + 8 * h + i;  // C: M = 8*(lane/16)+i
      const int bidx = m >> 10, t = m & (TSEQ - 1);
      const float val = __cosf(acc[j][i] + bb);
      int o;
      if (vmode)
        o = (((bidx * HEADS + head) * DK + d) * TSEQ + t);
      else
        o = (((bidx * HEADS + head) * TSEQ + t) * DK + d);
      out[o] = (_Float16)val;
    }
  }
}

// ---------------------------------------------------------------------------
// Output projection: Y = A @ Wo^T + bo, fp32 out.
// ---------------------------------------------------------------------------
__global__ void k_outproj(const _Float16* __restrict__ X,
                          const _Float16* __restrict__ W,
                          const float* __restrict__ bias,
                          float* __restrict__ out) {
  __shared__ __align__(16) _Float16 bt[2][64 * 32];
  GEMM_CORE(X, W, bt, acc)

#pragma unroll
  for (int j = 0; j < 4; ++j) {
    const int n = bn + j * 16 + r;
    const float bb = bias[n];
#pragma unroll
    for (int i = 0; i < 8; ++i) {
      const int m = bm * 128 + wave * 16 + 8 * h + i;
      out[m * EMBED + n] = acc[j][i] + bb;
    }
  }
}

// ---------------------------------------------------------------------------
// Flash attention: one wave per (b, head, 16-query tile); kv processed 32/iter.
//   Q,K: [b,h,t,16] f16;  Vt: [b,h,16,t] f16;  out: [b*t, EMBED] f16
// ---------------------------------------------------------------------------
#define AWAVES 4
__global__ void k_attn(const _Float16* __restrict__ Q,
                       const _Float16* __restrict__ K,
                       const _Float16* __restrict__ Vt,
                       _Float16* __restrict__ out) {
  __shared__ __align__(16) _Float16 lds[AWAVES][16 * 32];
  const int lane = threadIdx.x & 31;
  const int wave = threadIdx.x >> 5;
  int widx = blockIdx.x * AWAVES + wave;
  const int qt = widx & 63; widx >>= 6;        // 64 query tiles
  const int head = widx & 63; widx >>= 6;      // 64 heads
  const int b = widx;                          // batch
  const int r = lane & 15, h = lane >> 4;
  const int bh = b * HEADS + head;
  const _Float16* Qp = Q + bh * TSEQ * DK;
  const _Float16* Kp = K + bh * TSEQ * DK;
  const _Float16* Vp = Vt + bh * DK * TSEQ;
  _Float16* P = &lds[wave][0];

  // Q A-tile, d padded 16->32 with zeros
  v16h qa = h == 0 ? hcat(*(const v8h*)(Qp + (qt * 16 + r) * DK), *(const v8h*)(Qp + (qt * 16 + r) * DK + 8)) : hcat(hz8(), hz8());

  v8f o = fz8();
  float mrun[8], lrun[8];
#pragma unroll
  for (int i = 0; i < 8; ++i) { mrun[i] = -INFINITY; lrun[i] = 0.f; }

  const v16h zz = hcat(hz8(), hz8());
  for (int kv = 0; kv < TSEQ; kv += 32) {
    // K^T B-tiles (d padded 16->32: upper half-wave contributes zeros)
    v16h bk0, bk1;
    if (h == 0) {
      const _Float16* kp0 = Kp + (kv + r) * DK;
      const _Float16* kp1 = Kp + (kv + 16 + r) * DK;
      bk0 = hcat(*(const v8h*)kp0, *(const v8h*)(kp0 + 8));
      bk1 = hcat(*(const v8h*)kp1, *(const v8h*)(kp1 + 8));
    } else {
      bk0 = zz; bk1 = zz;
    }
    v8f s0 = wmma16(qa, bk0, fz8());   // scores cols kv..kv+15
    v8f s1 = wmma16(qa, bk1, fz8());   // scores cols kv+16..kv+31

    // online softmax over the 32 new kv columns (rows live across lanes%16)
#pragma unroll
    for (int i = 0; i < 8; ++i) {
      float a0 = s0[i] * 0.25f, a1 = s1[i] * 0.25f;   // 1/sqrt(16)
      float rm = fmaxf(a0, a1);
      rm = fmaxf(rm, __shfl_xor(rm, 1, 32));
      rm = fmaxf(rm, __shfl_xor(rm, 2, 32));
      rm = fmaxf(rm, __shfl_xor(rm, 4, 32));
      rm = fmaxf(rm, __shfl_xor(rm, 8, 32));
      const float nm = fmaxf(mrun[i], rm);
      const float alpha = __expf(mrun[i] - nm);
      const float p0 = __expf(a0 - nm);
      const float p1 = __expf(a1 - nm);
      float rs = p0 + p1;
      rs += __shfl_xor(rs, 1, 32);
      rs += __shfl_xor(rs, 2, 32);
      rs += __shfl_xor(rs, 4, 32);
      rs += __shfl_xor(rs, 8, 32);
      lrun[i] = lrun[i] * alpha + rs;
      mrun[i] = nm;
      o[i] *= alpha;
      // stage P (C layout -> LDS 16x32 row-major)
      P[(8 * h + i) * 32 + r] = (_Float16)p0;
      P[(8 * h + i) * 32 + 16 + r] = (_Float16)p1;
    }
    asm volatile("s_wait_dscnt 0" ::: "memory");
    // re-read P in A layout (full 16x32, no padding)
    v16h pa = hcat(*(const v8h*)(P + r * 32 + 8 * h),
                   *(const v8h*)(P + r * 32 + 16 + 8 * h));
    asm volatile("" ::: "memory");
    // V B-tile: lane n=r holds Vt[n, kv+8h .. kv+8h+7] and Vt[n, kv+16+8h .. kv+16+8h+7]
    const _Float16* vp = Vp + r * TSEQ + kv + 8 * h;
    v16h bv = hcat(*(const v8h*)vp, *(const v8h*)(vp + 16));
    o = wmma16(pa, bv, o);
  }

  // normalize and scatter to merged-head layout [b*t, EMBED]
#pragma unroll
  for (int i = 0; i < 8; ++i) {
    const int t = qt * 16 + 8 * h + i;
    out[(b * TSEQ + t) * EMBED + head * DK + r] = (_Float16)(o[i] / lrun[i]);
  }
}

// ---------------------------------------------------------------------------
extern "C" void kernel_launch(void* const* d_in, const int* in_sizes, int n_in,
                              void* d_out, int out_size, void* d_ws,
                              size_t ws_size, hipStream_t stream) {
  const float* x     = (const float*)d_in[0];
  const float* Wq    = (const float*)d_in[1];
  const float* bq    = (const float*)d_in[2];
  const float* Wk    = (const float*)d_in[3];
  const float* bk    = (const float*)d_in[4];
  const float* Wv    = (const float*)d_in[5];
  const float* bv    = (const float*)d_in[6];
  const float* Wo    = (const float*)d_in[7];
  const float* bo    = (const float*)d_in[8];
  const float* theta = (const float*)d_in[9];

  char* ws = (char*)d_ws;
  const size_t MB = 1u << 20;
  _Float16* Xh   = (_Float16*)(ws + 0 * MB);   // [2048,1024]        4 MiB
  _Float16* Wqh  = (_Float16*)(ws + 4 * MB);   // [1024,1024]        2 MiB
  _Float16* Wkh  = (_Float16*)(ws + 6 * MB);
  _Float16* Wvh  = (_Float16*)(ws + 8 * MB);
  _Float16* Woh  = (_Float16*)(ws + 10 * MB);
  _Float16* Qh   = (_Float16*)(ws + 12 * MB);  // [b,h,t,16]         4 MiB
  _Float16* Kh   = (_Float16*)(ws + 16 * MB);
  _Float16* Vt   = (_Float16*)(ws + 20 * MB);  // [b,h,16,t]
  _Float16* Attn = (_Float16*)(ws + 24 * MB);  // [2048,1024]        4 MiB

  // fp32 -> f16
  {
    int n4x = BT * EMBED / 4;
    k_cvt<<<(n4x + 255) / 256, 256, 0, stream>>>(x, Xh, n4x);
    int n4w = EMBED * EMBED / 4;
    k_cvt<<<(n4w + 255) / 256, 256, 0, stream>>>(Wq, Wqh, n4w);
    k_cvt<<<(n4w + 255) / 256, 256, 0, stream>>>(Wk, Wkh, n4w);
    k_cvt<<<(n4w + 255) / 256, 256, 0, stream>>>(Wv, Wvh, n4w);
    k_cvt<<<(n4w + 255) / 256, 256, 0, stream>>>(Wo, Woh, n4w);
  }

  // q/k/v projections + cos(.+theta); 256 blocks x 8 waves
  k_proj<<<256, 256, 0, stream>>>(Xh, Wqh, bq, theta, Qh, 0);
  k_proj<<<256, 256, 0, stream>>>(Xh, Wkh, bk, theta, Kh, 0);
  k_proj<<<256, 256, 0, stream>>>(Xh, Wvh, bv, theta, Vt, 1);

  // flash attention: 2*64*64 = 8192 wave-jobs, 4 waves/block
  k_attn<<<(BATCH * HEADS * (TSEQ / 16)) / AWAVES, AWAVES * 32, 0, stream>>>(
      Qh, Kh, Vt, Attn);

  // output projection, fp32 result
  k_outproj<<<256, 256, 0, stream>>>(Attn, Woh, bo, (float*)d_out);
}
